// GraphConvLayer_55018531062596
// MI455X (gfx1250) — matrix operand order repaired and checked by reading, not compiled, not run on hardware
//
#include <hip/hip_runtime.h>
#include <hip/hip_bf16.h>

// ---------------- problem constants ----------------
#define N_NODES   100000
#define N_EDGES   1600000
#define NODE_DIM  128
#define EDGE_DIM  32
#define HIDDEN    64
#define OUT_DIM   64

#define NT_NODE   (N_NODES / 16)   // 6250 node tiles
#define NT_EDGE   (N_EDGES / 16)   // 100000 edge tiles
#define WPB       8                // waves per block (256 threads, wave32)

// ---------------- workspace layout (bytes) ----------------
#define NC_OFF    ((size_t)0)
#define NP_OFF    (NC_OFF + (size_t)N_NODES * 64 * 4)
#define WB_OFF    (NP_OFF + (size_t)N_NODES * 64 * 4)
// bf16 B-tiles, each tile = 32 lanes * 16 el = 512 bf16 = 1024 B.
// tile index within a matrix region = kc*4 + nt  (kc = K/32 chunk, nt = N/16 tile)
#define T_WN      0    // Wn_fold  [128,64] -> 16 tiles
#define T_WU1N    16   // wu1_nf   [128,64] -> 16 tiles
#define T_WE      32   // We_fold  [ 32,64] ->  4 tiles
#define T_WM2     36   // wm2      [ 64,64] ->  8 tiles
#define T_WU1A    44   // wu1_agg  [ 64,64] ->  8 tiles
#define T_WU2     52   // wu2      [ 64,64] ->  8 tiles
#define NW_TILES  60
#define BIAS_OFF  (WB_OFF + (size_t)NW_TILES * 512 * 2)

typedef __attribute__((ext_vector_type(16))) __bf16 v16bf;
typedef __attribute__((ext_vector_type(8)))  float  v8f;

// ---------------- WMMA helpers ----------------
__device__ __forceinline__ v8f wmma_bf16(v16bf a, v16bf b, v8f c) {
  return __builtin_amdgcn_wmma_f32_16x16x32_bf16(false, a, false, b, (short)0, c,
                                                 false, false);
}

// B tile from LDS (pre-swizzled: contiguous 16 bf16 per lane)
__device__ __forceinline__ v16bf load_btile_lds(const __bf16* lw, int tile, int lane) {
  return *(const v16bf*)(lw + ((unsigned)tile * 32 + lane) * 16);
}

// Cooperative copy of `n_bf16` bf16 weights (multiple of 8) global -> LDS.
__device__ __forceinline__ void stage_weights(const __bf16* __restrict__ src,
                                              __bf16* dst, int n_bf16) {
  const int n16 = n_bf16 >> 3;           // uint4 = 16 B = 8 bf16
  for (int i = threadIdx.x; i < n16; i += blockDim.x)
    ((uint4*)dst)[i] = ((const uint4*)src)[i];
  __syncthreads();
}

// Build an A tile (16x32 bf16) for this lane from a row-major fp32 row.
// A layout (matches B packing): lanes 0-15 (row M=lane): el 0..15 = K 0..15.
//               lanes 16-31 (row M=lane-16): el 0..15 = K 16..31.  kb = hi*8.
__device__ __forceinline__ v16bf make_a_chunk(const float* chunk, int kb) {
  float4 x0 = *(const float4*)(chunk + 2 * kb);
  float4 x1 = *(const float4*)(chunk + 2 * kb + 4);
  float4 x2 = *(const float4*)(chunk + 2 * kb + 8);
  float4 x3 = *(const float4*)(chunk + 2 * kb + 12);
  v16bf a;
  a[0]  = (__bf16)x0.x; a[1]  = (__bf16)x0.y; a[2]  = (__bf16)x0.z; a[3]  = (__bf16)x0.w;
  a[4]  = (__bf16)x1.x; a[5]  = (__bf16)x1.y; a[6]  = (__bf16)x1.z; a[7]  = (__bf16)x1.w;
  a[8]  = (__bf16)x2.x; a[9]  = (__bf16)x2.y; a[10] = (__bf16)x2.z; a[11] = (__bf16)x2.w;
  a[12] = (__bf16)x3.x; a[13] = (__bf16)x3.y; a[14] = (__bf16)x3.z; a[15] = (__bf16)x3.w;
  return a;
}

// ---------------- kernel 0: fold + bf16-swizzle weights ----------------
// B layout (K=32 bf16): n = nt*16 + (lane&15), k = kc*32 + (lane>=16?16:0) + i.
__global__ void fold_weights(const float* __restrict__ wn,  const float* __restrict__ bn,
                             const float* __restrict__ we,  const float* __restrict__ be,
                             const float* __restrict__ wm1, const float* __restrict__ bm1,
                             const float* __restrict__ wm2, const float* __restrict__ bm2,
                             const float* __restrict__ wu1, const float* __restrict__ bu1,
                             const float* __restrict__ wu2, const float* __restrict__ bu2,
                             __bf16* __restrict__ wsb, float* __restrict__ biases) {
  const int tid = threadIdx.x;

  // Wn_fold = wn @ wm1[0:64,:]   (K = 128)
  for (int slot = tid; slot < 128 * 64; slot += blockDim.x) {
    int i = slot & 15, lane = (slot >> 4) & 31, t = slot >> 9;
    int k = (t >> 2) * 32 + ((lane & 16) ? 16 : 0) + i;
    int n = (t & 3) * 16 + (lane & 15);
    float v = 0.f;
    for (int h = 0; h < HIDDEN; ++h) v += wn[k * HIDDEN + h] * wm1[h * HIDDEN + n];
    wsb[(size_t)T_WN * 512 + slot] = (__bf16)v;
  }
  // wu1_nf = wu1[0:128,:]        (K = 128, plain convert)
  for (int slot = tid; slot < 128 * 64; slot += blockDim.x) {
    int i = slot & 15, lane = (slot >> 4) & 31, t = slot >> 9;
    int k = (t >> 2) * 32 + ((lane & 16) ? 16 : 0) + i;
    int n = (t & 3) * 16 + (lane & 15);
    wsb[(size_t)T_WU1N * 512 + slot] = (__bf16)wu1[k * HIDDEN + n];
  }
  // We_fold = we @ wm1[64:128,:] (K = 32)
  for (int slot = tid; slot < 32 * 64; slot += blockDim.x) {
    int i = slot & 15, lane = (slot >> 4) & 31, t = slot >> 9;
    int k = (t >> 2) * 32 + ((lane & 16) ? 16 : 0) + i;
    int n = (t & 3) * 16 + (lane & 15);
    float v = 0.f;
    for (int h = 0; h < HIDDEN; ++h)
      v += we[k * HIDDEN + h] * wm1[(HIDDEN + h) * HIDDEN + n];
    wsb[(size_t)T_WE * 512 + slot] = (__bf16)v;
  }
  // wm2                           (K = 64)
  for (int slot = tid; slot < 64 * 64; slot += blockDim.x) {
    int i = slot & 15, lane = (slot >> 4) & 31, t = slot >> 9;
    int k = (t >> 2) * 32 + ((lane & 16) ? 16 : 0) + i;
    int n = (t & 3) * 16 + (lane & 15);
    wsb[(size_t)T_WM2 * 512 + slot] = (__bf16)wm2[k * OUT_DIM + n];
  }
  // wu1_agg = wu1[128:192,:]      (K = 64)
  for (int slot = tid; slot < 64 * 64; slot += blockDim.x) {
    int i = slot & 15, lane = (slot >> 4) & 31, t = slot >> 9;
    int k = (t >> 2) * 32 + ((lane & 16) ? 16 : 0) + i;
    int n = (t & 3) * 16 + (lane & 15);
    wsb[(size_t)T_WU1A * 512 + slot] = (__bf16)wu1[(NODE_DIM + k) * HIDDEN + n];
  }
  // wu2                           (K = 64)
  for (int slot = tid; slot < 64 * 64; slot += blockDim.x) {
    int i = slot & 15, lane = (slot >> 4) & 31, t = slot >> 9;
    int k = (t >> 2) * 32 + ((lane & 16) ? 16 : 0) + i;
    int n = (t & 3) * 16 + (lane & 15);
    wsb[(size_t)T_WU2 * 512 + slot] = (__bf16)wu2[k * OUT_DIM + n];
  }
  // biases: b_msg = bm1 + be@wm1_e + bn@wm1_n ; bm2 ; bu1 ; bu2
  for (int j = tid; j < 64; j += blockDim.x) {
    float v = bm1[j];
    for (int h = 0; h < HIDDEN; ++h) v += be[h] * wm1[(HIDDEN + h) * HIDDEN + j];
    for (int h = 0; h < HIDDEN; ++h) v += bn[h] * wm1[h * HIDDEN + j];
    biases[j]       = v;
    biases[64 + j]  = bm2[j];
    biases[128 + j] = bu1[j];
    biases[192 + j] = bu2[j];
  }
}

// ---------------- zero fill (aggregated buffer == d_out) ----------------
__global__ void zero_f32(float* __restrict__ p, size_t n) {
  size_t i = (size_t)blockIdx.x * blockDim.x + threadIdx.x;
  size_t stride = (size_t)gridDim.x * blockDim.x;
  for (; i < n; i += stride) p[i] = 0.f;
}

// ---------------- kernel 1: per-node precompute ----------------
// node_contrib = NF @ Wn_fold ; node_pre = NF @ wu1_nf
// Two sequential passes to halve peak register pressure.
__global__ void __launch_bounds__(256, 1)
node_precompute(const float* __restrict__ nf,
                const __bf16* __restrict__ wsb,
                float* __restrict__ nc, float* __restrict__ np) {
  __shared__ __align__(32) __bf16 lw[32 * 512];   // 32 tiles (T_WN .. T_WU1N+15), 32 KB
  stage_weights(wsb + (size_t)T_WN * 512, lw, 32 * 512);

  const int lane = threadIdx.x & 31;
  const int wv   = threadIdx.x >> 5;
  const int hi   = lane >> 4;
  const int nlo  = lane & 15;
  const size_t wave = (size_t)blockIdx.x * WPB + wv;
  const size_t nwaves = (size_t)gridDim.x * WPB;

  for (size_t t = wave; t < NT_NODE; t += nwaves) {
    const float* arow = nf + (t * 16 + nlo) * NODE_DIM;

    // pass 1: node_contrib
    {
      v8f acc[4] = {{}, {}, {}, {}};
#pragma unroll
      for (int kc = 0; kc < 4; ++kc) {
        v16bf a = make_a_chunk(arow + kc * 32, hi * 8);
#pragma unroll
        for (int nt = 0; nt < 4; ++nt)
          acc[nt] = wmma_bf16(a, load_btile_lds(lw, kc * 4 + nt, lane), acc[nt]);
      }
#pragma unroll
      for (int nt = 0; nt < 4; ++nt) {
        int n = nt * 16 + nlo;
#pragma unroll
        for (int v = 0; v < 8; ++v)
          nc[(t * 16 + v + hi * 8) * 64 + n] = acc[nt][v];
      }
    }
    // pass 2: node_pre
    {
      v8f acc[4] = {{}, {}, {}, {}};
#pragma unroll
      for (int kc = 0; kc < 4; ++kc) {
        v16bf a = make_a_chunk(arow + kc * 32, hi * 8);
#pragma unroll
        for (int nt = 0; nt < 4; ++nt)
          acc[nt] = wmma_bf16(a, load_btile_lds(lw, 16 + kc * 4 + nt, lane), acc[nt]);
      }
#pragma unroll
      for (int nt = 0; nt < 4; ++nt) {
        int n = nt * 16 + nlo;
#pragma unroll
        for (int v = 0; v < 8; ++v)
          np[(t * 16 + v + hi * 8) * 64 + n] = acc[nt][v];
      }
    }
  }
}

// ---------------- kernel 2: edge message + scatter ----------------
__global__ void __launch_bounds__(256, 1)
edge_kernel(const float* __restrict__ ef,
            const long long* __restrict__ eidx,
            const float* __restrict__ nc,
            const __bf16* __restrict__ wsb,
            const float* __restrict__ biases,
            float* __restrict__ agg) {
  __shared__ float smem_tr[WPB * 16 * 64];        // 32 KB transpose buffers
  __shared__ __align__(32) __bf16 lw[12 * 512];   // We_fold(4) + wm2(8), 12 KB
  stage_weights(wsb + (size_t)T_WE * 512, lw, 12 * 512);

  const int lane = threadIdx.x & 31;
  const int wv   = threadIdx.x >> 5;
  const int hi   = lane >> 4;
  const int nlo  = lane & 15;
  float* my = smem_tr + wv * (16 * 64);

  float bmsg[4], bm2r[4];
#pragma unroll
  for (int nt = 0; nt < 4; ++nt) {
    bmsg[nt] = biases[nt * 16 + nlo];
    bm2r[nt] = biases[64 + nt * 16 + nlo];
  }

  // node ids are < 2^31: read only the low dword of each int64 index
  const int* srcp = (const int*)eidx;                 // [2*e]
  const int* dstp = (const int*)(eidx + N_EDGES);     // [2*e]
  const size_t wave = (size_t)blockIdx.x * WPB + wv;
  const size_t nwaves = (size_t)gridDim.x * WPB;

  for (size_t t = wave; t < NT_EDGE; t += nwaves) {
    if (t + nwaves < NT_EDGE)
      __builtin_prefetch(ef + ((t + nwaves) * 16 + nlo) * EDGE_DIM, 0, 0);

    // A tile: 16 edges x 32 edge-feature dims (fp32 -> bf16)
    const float* arow = ef + (t * 16 + nlo) * EDGE_DIM;
    v16bf a = make_a_chunk(arow, hi * 8);

    int sidx[8], didx[8];   // per-lane src/dst for its 8 C-rows (M = v + 8*hi)
#pragma unroll
    for (int v = 0; v < 8; ++v) {
      sidx[v] = srcp[2 * (t * 16 + v + hi * 8)];
      didx[v] = dstp[2 * (t * 16 + v + hi * 8)];
    }

    // GEMM1: EF @ We_fold ; + gathered node_contrib[src] + b_msg ; ReLU -> LDS
#pragma unroll
    for (int nt = 0; nt < 4; ++nt) {
      v8f c = {};
      c = wmma_bf16(a, load_btile_lds(lw, nt, lane), c);
      int n = nt * 16 + nlo;
#pragma unroll
      for (int v = 0; v < 8; ++v) {
        float g = nc[(size_t)sidx[v] * 64 + n];         // L2-resident gather
        float x = c[v] + g + bmsg[nt];
        my[(v + hi * 8) * 64 + n] = fmaxf(x, 0.f);      // D-layout -> row-major LDS
      }
    }
    asm volatile("s_wait_dscnt 0" ::: "memory");        // LDS RAW fence (same wave)

    // reload in A layout (row M = lane&15), two K=32 chunks
    v16bf a2_0 = make_a_chunk(my + nlo * 64 + 0,  hi * 8);
    v16bf a2_1 = make_a_chunk(my + nlo * 64 + 32, hi * 8);

    // GEMM2: m1 @ wm2 + bm2, then scatter-add into aggregated[dst]
#pragma unroll
    for (int nt = 0; nt < 4; ++nt) {
      v8f c = {};
      c = wmma_bf16(a2_0, load_btile_lds(lw, 4 + nt, lane), c);
      c = wmma_bf16(a2_1, load_btile_lds(lw, 8 + nt, lane), c);
      int n = nt * 16 + nlo;
#pragma unroll
      for (int v = 0; v < 8; ++v)
        unsafeAtomicAdd(agg + (size_t)didx[v] * 64 + n, c[v] + bm2r[nt]);
    }
  }
}

// ---------------- kernel 3: node update (in place over d_out) ----------------
__global__ void __launch_bounds__(256, 1)
node_update(const float* agg_in,            // == out (aliased)
            const float* __restrict__ np,
            const __bf16* __restrict__ wsb,
            const float* __restrict__ biases,
            float* out) {
  __shared__ float smem_tr[WPB * 16 * 64];        // 32 KB
  __shared__ __align__(32) __bf16 lw[16 * 512];   // wu1_agg(8) + wu2(8), 16 KB
  stage_weights(wsb + (size_t)T_WU1A * 512, lw, 16 * 512);

  const int lane = threadIdx.x & 31;
  const int wv   = threadIdx.x >> 5;
  const int hi   = lane >> 4;
  const int nlo  = lane & 15;
  float* my = smem_tr + wv * (16 * 64);

  float bu1r[4], bu2r[4];
#pragma unroll
  for (int nt = 0; nt < 4; ++nt) {
    bu1r[nt] = biases[128 + nt * 16 + nlo];
    bu2r[nt] = biases[192 + nt * 16 + nlo];
  }

  const size_t wave = (size_t)blockIdx.x * WPB + wv;
  const size_t nwaves = (size_t)gridDim.x * WPB;

  for (size_t t = wave; t < NT_NODE; t += nwaves) {
    const float* arow = agg_in + (t * 16 + nlo) * 64;    // read agg rows first
    v16bf a0 = make_a_chunk(arow + 0,  hi * 8);
    v16bf a1 = make_a_chunk(arow + 32, hi * 8);

    // h = relu(agg @ wu1_agg + node_pre + bu1) -> LDS
#pragma unroll
    for (int nt = 0; nt < 4; ++nt) {
      v8f c = {};
      c = wmma_bf16(a0, load_btile_lds(lw, nt, lane), c);
      c = wmma_bf16(a1, load_btile_lds(lw, 4 + nt, lane), c);
      int n = nt * 16 + nlo;
#pragma unroll
      for (int v = 0; v < 8; ++v) {
        size_t node = t * 16 + v + hi * 8;
        float x = c[v] + np[node * 64 + n] + bu1r[nt];
        my[(v + hi * 8) * 64 + n] = fmaxf(x, 0.f);
      }
    }
    asm volatile("s_wait_dscnt 0" ::: "memory");

    v16bf h0 = make_a_chunk(my + nlo * 64 + 0,  hi * 8);
    v16bf h1 = make_a_chunk(my + nlo * 64 + 32, hi * 8);

    // out = h @ wu2 + bu2  (overwrites this tile's rows of the aliased buffer)
#pragma unroll
    for (int nt = 0; nt < 4; ++nt) {
      v8f c = {};
      c = wmma_bf16(h0, load_btile_lds(lw, 8 + nt, lane), c);
      c = wmma_bf16(h1, load_btile_lds(lw, 12 + nt, lane), c);
      int n = nt * 16 + nlo;
#pragma unroll
      for (int v = 0; v < 8; ++v)
        out[(t * 16 + v + hi * 8) * 64 + n] = c[v] + bu2r[nt];
    }
  }
}

// ---------------- launcher ----------------
extern "C" void kernel_launch(void* const* d_in, const int* in_sizes, int n_in,
                              void* d_out, int out_size, void* d_ws, size_t ws_size,
                              hipStream_t stream) {
  const float*     nf   = (const float*)d_in[0];
  const long long* eidx = (const long long*)d_in[1];
  const float*     ef   = (const float*)d_in[2];
  const float*     wn   = (const float*)d_in[3];
  const float*     bn   = (const float*)d_in[4];
  const float*     we   = (const float*)d_in[5];
  const float*     be   = (const float*)d_in[6];
  const float*     wm1  = (const float*)d_in[7];
  const float*     bm1  = (const float*)d_in[8];
  const float*     wm2  = (const float*)d_in[9];
  const float*     bm2  = (const float*)d_in[10];
  const float*     wu1  = (const float*)d_in[11];
  const float*     bu1  = (const float*)d_in[12];
  const float*     wu2  = (const float*)d_in[13];
  const float*     bu2  = (const float*)d_in[14];

  float*  out    = (float*)d_out;
  char*   ws     = (char*)d_ws;
  float*  nc     = (float*)(ws + NC_OFF);
  float*  np     = (float*)(ws + NP_OFF);
  __bf16* wsb    = (__bf16*)(ws + WB_OFF);
  float*  biases = (float*)(ws + BIAS_OFF);

  fold_weights<<<1, 256, 0, stream>>>(wn, bn, we, be, wm1, bm1, wm2, bm2,
                                      wu1, bu1, wu2, bu2, wsb, biases);
  zero_f32<<<1024, 256, 0, stream>>>(out, (size_t)N_NODES * OUT_DIM);
  node_precompute<<<(NT_NODE + WPB - 1) / WPB, 256, 0, stream>>>(nf, wsb, nc, np);
  edge_kernel<<<2048, 256, 0, stream>>>(ef, eidx, nc, wsb, biases, out);
  node_update<<<(NT_NODE + WPB - 1) / WPB, 256, 0, stream>>>(out, np, wsb, biases, out);
}
